// MultiHeadAttention_5136780886302
// MI455X (gfx1250) — matrix operand order repaired and checked by reading, not compiled, not run on hardware
//
#include <hip/hip_runtime.h>
#include <hip/hip_bf16.h>

// MI455X / gfx1250: wave32, WMMA 16x16x32 bf16 (f32 accumulate) for all GEMMs,
// gfx1250 async-to-LDS (GLOBAL_LOAD_ASYNC_TO_LDS_B128 + s_wait_asynccnt) for
// double-buffered tile staging.

typedef __attribute__((ext_vector_type(16))) __bf16 v16bf;
typedef __attribute__((ext_vector_type(8)))  float  v8f;

union FragAB { v16bf v; uint4 u[2]; };

__device__ __forceinline__ unsigned short f2bf(float f) {
  unsigned int u = __float_as_uint(f);
  u += 0x7FFFu + ((u >> 16) & 1u);        // round-to-nearest-even
  return (unsigned short)(u >> 16);
}

// gfx1250 async copy: per-lane 16B global -> LDS, tracked by ASYNCcnt.
// lds_off = workgroup-relative LDS byte address (low 32 bits of generic ptr).
__device__ __forceinline__ void async_ld_b128(unsigned lds_off,
                                              unsigned long long gaddr) {
  asm volatile("global_load_async_to_lds_b128 %0, %1, off"
               :: "v"(lds_off), "v"(gaddr) : "memory");
}
__device__ __forceinline__ unsigned lds_addr32(const void* p) {
  return (unsigned)(unsigned long long)p;
}
#define WAIT_ASYNC(n) asm volatile("s_wait_asynccnt " #n ::: "memory")

// ---------------------------------------------------------------- converts
__global__ void cvt_bf16(const float* __restrict__ in,
                         unsigned short* __restrict__ out, int n) {
  int i = blockIdx.x * blockDim.x + threadIdx.x;
  if (i < n) out[i] = f2bf(in[i]);
}

// in: [1024 K x 1024 N] fp32 row-major -> out: [1024 N x 1024 K] bf16
__global__ void cvt_w_t(const float* __restrict__ in,
                        unsigned short* __restrict__ out) {
  int i = blockIdx.x * blockDim.x + threadIdx.x;   // 1M threads
  int n = i >> 10, k = i & 1023;
  out[i] = f2bf(in[k * 1024 + n]);
}

// ---------------------------------------------------------------- GEMM
// C[4096 x 1024] = A[4096 x 1024] * B, Bt given as [N x K] bf16.
// Block: 256 thr = 8 waves (4 along M x 2 along N). Block tile 128M x 128N,
// wave tile 32M x 64N (8 f32 accumulators). A and B k-slabs (128x32 bf16)
// staged via async-to-LDS, double buffered.
// mode 0: out bf16 -> [B,H,S,HD]   (Q, K)
// mode 1: out bf16 -> [B,H,HD,S]   (V, pre-transposed for P@V B-operand)
// mode 2: out fp32 -> row-major, + bias + residual (pre-LayerNorm buffer)
__global__ __launch_bounds__(256) void gemm_bf16(
    const unsigned short* __restrict__ A,
    const unsigned short* __restrict__ Bt,
    const float* __restrict__ bias,
    const float* __restrict__ resid,
    unsigned short* __restrict__ outb,
    float* __restrict__ outf,
    int mode) {
  __shared__ alignas(16) unsigned short Abuf[2][128 * 32];  // 2 x 8 KB
  __shared__ alignas(16) unsigned short Bbuf[2][128 * 32];  // 2 x 8 KB
  const int tid  = threadIdx.x;
  const int lane = tid & 31, wave = tid >> 5;
  const int half = lane >> 4, l16 = lane & 15;
  const int waveM = wave >> 1, waveN = wave & 1;
  const int m0 = blockIdx.y * 128;
  const int n0 = blockIdx.x * 128;

  const unsigned aBase = lds_addr32(&Abuf[0][0]);
  const unsigned bBase = lds_addr32(&Bbuf[0][0]);

  v8f acc[2][4] = {};

  // stage one 128x32 k-slab of A and Bt: 512 chunks of 16B each, 2/thread/mat.
  // linear chunk c -> row = c>>2, 16B-seg = c&3 (row-major [128][32] layout).
  auto stage = [&](int k0, int buf) {
#pragma unroll
    for (int q = 0; q < 2; ++q) {
      const int c = tid + q * 256;
      const unsigned loff = (unsigned)(buf * 8192 + c * 16);
      async_ld_b128(aBase + loff,
          (unsigned long long)(A  + (size_t)(m0 + (c >> 2)) * 1024 + (c & 3) * 8 + k0));
      async_ld_b128(bBase + loff,
          (unsigned long long)(Bt + (size_t)(n0 + (c >> 2)) * 1024 + (c & 3) * 8 + k0));
    }
  };

  stage(0, 0);
  for (int ks = 0; ks < 32; ++ks) {
    const int cur = ks & 1;
    if (ks < 31) {
      stage((ks + 1) * 32, cur ^ 1);
      WAIT_ASYNC(4);    // 4 newest in flight = next stage; current is resident
    } else {
      WAIT_ASYNC(0);
    }
    __syncthreads();

    const unsigned short* As = &Abuf[cur][0];
    const unsigned short* Bs = &Bbuf[cur][0];

    // A fragments (ISA 16-bit A layout: lane<16 K={0..7,16..23}, lane>=16 +8)
    FragAB fa[2];
#pragma unroll
    for (int mt = 0; mt < 2; ++mt) {
      const unsigned short* ar = As + (waveM * 32 + mt * 16 + l16) * 32 + half * 8;
      fa[mt].u[0] = *(const uint4*)(ar);
      fa[mt].u[1] = *(const uint4*)(ar + 16);
    }
#pragma unroll
    for (int nt = 0; nt < 4; ++nt) {
      FragAB fb;  // B fragment: lane = col N, K={0..7,16..23} (+8 for half 1)
      const unsigned short* br = Bs + (waveN * 64 + nt * 16 + l16) * 32 + half * 8;
      fb.u[0] = *(const uint4*)(br);
      fb.u[1] = *(const uint4*)(br + 16);
      acc[0][nt] = __builtin_amdgcn_wmma_f32_16x16x32_bf16(
          false, fa[0].v, false, fb.v, (short)0, acc[0][nt], false, false);
      acc[1][nt] = __builtin_amdgcn_wmma_f32_16x16x32_bf16(
          false, fa[1].v, false, fb.v, (short)0, acc[1][nt], false, false);
    }
    __syncthreads();   // protect the buffer the next stage() will overwrite
  }

  // epilogue: C layout -> component i is row (half*8+i), lane&15 is col
#pragma unroll
  for (int mt = 0; mt < 2; ++mt) {
#pragma unroll
    for (int nt = 0; nt < 4; ++nt) {
      const int c = n0 + waveN * 64 + nt * 16 + l16;
      const float bv = bias[c];
#pragma unroll
      for (int i = 0; i < 8; ++i) {
        const int r = m0 + waveM * 32 + mt * 16 + half * 8 + i;
        float val = acc[mt][nt][i] + bv;
        if (mode == 2) {
          const size_t idx = (size_t)r * 1024 + c;
          outf[idx] = val + resid[idx];
        } else {
          const int b = r >> 11, s = r & 2047;
          const int h = c >> 6,  hd = c & 63;
          size_t idx;
          if (mode == 0) idx = (((size_t)(b * 16 + h) * 2048 + s) * 64 + hd);
          else           idx = (((size_t)(b * 16 + h) * 64 + hd) * 2048 + s);
          outb[idx] = f2bf(val);
        }
      }
    }
  }
}

// ---------------------------------------------------------------- attention
// One wave per 16-query tile; 8 waves per block share one head's K/V stream.
// K (32x64) and Vt (64x32) j-tiles staged per block via async-to-LDS
// (double buffered); P transposed C-layout -> A-layout through per-wave LDS.
// Q: [BH,S,64] bf16, K: [BH,S,64] bf16, Vt: [BH,64,S] bf16, O: [B,S,1024] bf16
__global__ __launch_bounds__(256) void attention_fa(
    const unsigned short* __restrict__ Q,
    const unsigned short* __restrict__ Km,
    const unsigned short* __restrict__ Vt,
    unsigned short* __restrict__ O) {
  __shared__ alignas(16) unsigned short Kbuf[2][32 * 64];   // 2 x 4 KB
  __shared__ alignas(16) unsigned short Vbuf[2][64 * 32];   // 2 x 4 KB
  __shared__ alignas(16) unsigned short ldsP[8][16 * 32];   // per-wave P tile
  const int tid  = threadIdx.x;
  const int lane = tid & 31, wave = tid >> 5;
  const int half = lane >> 4, l16 = lane & 15;
  const int bh = blockIdx.y;
  const int q0 = (blockIdx.x * 8 + wave) * 16;

  const unsigned short* Qb = Q  + (size_t)bh * 2048 * 64;
  const unsigned short* Kb = Km + (size_t)bh * 2048 * 64;
  const unsigned short* Vb = Vt + (size_t)bh * 64 * 2048;
  const unsigned kBase = lds_addr32(&Kbuf[0][0]);
  const unsigned vBase = lds_addr32(&Vbuf[0][0]);

  // stage 32-key step: K tile rows are keys (stride 64), V tile rows are hd
  // (stride 32). 256 chunks of 16B per matrix, 1 per thread per matrix.
  auto stage = [&](int j, int buf) {
    async_ld_b128(kBase + (unsigned)(buf * 4096 + tid * 16),
        (unsigned long long)(Kb + (size_t)(j + (tid >> 3)) * 64 + (tid & 7) * 8));
    async_ld_b128(vBase + (unsigned)(buf * 4096 + tid * 16),
        (unsigned long long)(Vb + (size_t)(tid >> 2) * 2048 + j + (tid & 3) * 8));
  };

  // Q fragments: A-operand 16x32, two hd-chunks, resident in VGPRs
  FragAB fq[2];
#pragma unroll
  for (int c = 0; c < 2; ++c) {
    const unsigned short* p = Qb + (size_t)(q0 + l16) * 64 + c * 32 + half * 8;
    fq[c].u[0] = *(const uint4*)(p);
    fq[c].u[1] = *(const uint4*)(p + 16);
  }

  v8f accO[4] = {};
  float mrun[8], lrun[8];
#pragma unroll
  for (int i = 0; i < 8; ++i) { mrun[i] = -1e30f; lrun[i] = 0.f; }
  const float scale = 0.125f;   // 1/sqrt(64)

  stage(0, 0);
  for (int jj = 0; jj < 64; ++jj) {
    const int cur = jj & 1;
    if (jj < 63) {
      stage((jj + 1) * 32, cur ^ 1);
      WAIT_ASYNC(2);
    } else {
      WAIT_ASYNC(0);
    }
    __syncthreads();
    const unsigned short* Ks = &Kbuf[cur][0];
    const unsigned short* Vs = &Vbuf[cur][0];

    // scores: two 16x16 key tiles, contraction over hd=64 (2 chunks)
    v8f sc[2] = {};
#pragma unroll
    for (int t = 0; t < 2; ++t) {
#pragma unroll
      for (int c = 0; c < 2; ++c) {
        FragAB fk;   // B-operand = K^T: lane = key col, hd={0..7,16..23} (+8 for half 1)
        const unsigned short* p = Ks + (t * 16 + l16) * 64 + c * 32 + half * 8;
        fk.u[0] = *(const uint4*)(p);
        fk.u[1] = *(const uint4*)(p + 16);
        sc[t] = __builtin_amdgcn_wmma_f32_16x16x32_bf16(
            false, fq[c].v, false, fk.v, (short)0, sc[t], false, false);
      }
    }

    // online softmax; rows live in accumulator components, reductions are
    // 16-lane butterflies (masks 1,2,4,8 stay within each wave32 half)
    unsigned short* pw = &ldsP[wave][0];
#pragma unroll
    for (int i = 0; i < 8; ++i) {
      float s0 = sc[0][i] * scale, s1 = sc[1][i] * scale;
      float mt = fmaxf(s0, s1);
      mt = fmaxf(mt, __shfl_xor(mt, 1, 32));
      mt = fmaxf(mt, __shfl_xor(mt, 2, 32));
      mt = fmaxf(mt, __shfl_xor(mt, 4, 32));
      mt = fmaxf(mt, __shfl_xor(mt, 8, 32));
      float mnew  = fmaxf(mrun[i], mt);
      float alpha = __expf(mrun[i] - mnew);
      float p0 = __expf(s0 - mnew), p1 = __expf(s1 - mnew);
      float rs = p0 + p1;
      rs += __shfl_xor(rs, 1, 32);
      rs += __shfl_xor(rs, 2, 32);
      rs += __shfl_xor(rs, 4, 32);
      rs += __shfl_xor(rs, 8, 32);
      lrun[i] = lrun[i] * alpha + rs;
      mrun[i] = mnew;
      const int row = half * 8 + i;            // C-layout -> LDS transpose
      pw[row * 32 + l16]      = f2bf(p0);
      pw[row * 32 + 16 + l16] = f2bf(p1);
#pragma unroll
      for (int ht = 0; ht < 4; ++ht) accO[ht][i] *= alpha;
    }

    // P back as A-operand fragment (2x ds_load_b128, per-wave LDS, in-order)
    FragAB fp;
    const unsigned short* pr = pw + l16 * 32 + half * 8;
    fp.u[0] = *(const uint4*)(pr);
    fp.u[1] = *(const uint4*)(pr + 16);

#pragma unroll
    for (int ht = 0; ht < 4; ++ht) {
      FragAB fv;   // B-operand = V: lane = hd col; kv={0..7,16..23} (+8 for half 1)
      const unsigned short* vp = Vs + (ht * 16 + l16) * 32 + half * 8;
      fv.u[0] = *(const uint4*)(vp);
      fv.u[1] = *(const uint4*)(vp + 16);
      accO[ht] = __builtin_amdgcn_wmma_f32_16x16x32_bf16(
          false, fp.v, false, fv.v, (short)0, accO[ht], false, false);
    }
    __syncthreads();
  }

  // normalize + write [B,S,D] bf16
  const int b = bh >> 4, h = bh & 15;
#pragma unroll
  for (int ht = 0; ht < 4; ++ht) {
    const int col = h * 64 + ht * 16 + l16;
#pragma unroll
    for (int i = 0; i < 8; ++i) {
      const int row = b * 2048 + q0 + half * 8 + i;
      O[(size_t)row * 1024 + col] = f2bf(accO[ht][i] / lrun[i]);
    }
  }
}

// ---------------------------------------------------------------- layernorm
__global__ __launch_bounds__(256) void layernorm_row(
    const float* __restrict__ y, const float* __restrict__ gamma,
    const float* __restrict__ beta, float* __restrict__ out) {
  __shared__ float red[256];
  const int r = blockIdx.x;
  const float* yr = y + (size_t)r * 1024;
  float s = 0.f;
  for (int c = threadIdx.x; c < 1024; c += 256) s += yr[c];
  red[threadIdx.x] = s; __syncthreads();
  for (int w = 128; w > 0; w >>= 1) {
    if (threadIdx.x < w) red[threadIdx.x] += red[threadIdx.x + w];
    __syncthreads();
  }
  const float mu = red[0] * (1.f / 1024.f);
  __syncthreads();
  float v = 0.f;
  for (int c = threadIdx.x; c < 1024; c += 256) {
    float d = yr[c] - mu; v += d * d;
  }
  red[threadIdx.x] = v; __syncthreads();
  for (int w = 128; w > 0; w >>= 1) {
    if (threadIdx.x < w) red[threadIdx.x] += red[threadIdx.x + w];
    __syncthreads();
  }
  const float rstd = rsqrtf(red[0] * (1.f / 1024.f) + 1e-5f);
  for (int c = threadIdx.x; c < 1024; c += 256)
    out[(size_t)r * 1024 + c] = (yr[c] - mu) * rstd * gamma[c] + beta[c];
}

// ---------------------------------------------------------------- launch
extern "C" void kernel_launch(void* const* d_in, const int* in_sizes, int n_in,
                              void* d_out, int out_size, void* d_ws,
                              size_t ws_size, hipStream_t stream) {
  const float* x     = (const float*)d_in[0];
  const float* Wq    = (const float*)d_in[1];
  const float* bq    = (const float*)d_in[2];
  const float* Wk    = (const float*)d_in[3];
  const float* bk    = (const float*)d_in[4];
  const float* Wv    = (const float*)d_in[5];
  const float* bv    = (const float*)d_in[6];
  const float* Wp    = (const float*)d_in[7];
  const float* bp    = (const float*)d_in[8];
  const float* gamma = (const float*)d_in[9];
  const float* beta  = (const float*)d_in[10];
  float* out = (float*)d_out;

  // workspace layout (64 MB total)
  char* ws = (char*)d_ws;
  const size_t MB = 1024 * 1024;
  unsigned short* xb  = (unsigned short*)(ws + 0);        //  8 MB x bf16
  unsigned short* WqT = (unsigned short*)(ws + 8  * MB);  //  2 MB
  unsigned short* WkT = (unsigned short*)(ws + 10 * MB);  //  2 MB
  unsigned short* WvT = (unsigned short*)(ws + 12 * MB);  //  2 MB
  unsigned short* WpT = (unsigned short*)(ws + 14 * MB);  //  2 MB
  unsigned short* Qh  = (unsigned short*)(ws + 16 * MB);  //  8 MB [BH,S,HD]
  unsigned short* Kh  = (unsigned short*)(ws + 24 * MB);  //  8 MB [BH,S,HD]
  unsigned short* Vt  = (unsigned short*)(ws + 32 * MB);  //  8 MB [BH,HD,S]
  unsigned short* AO  = (unsigned short*)(ws + 40 * MB);  //  8 MB [B,S,D]
  float*          Yt  = (float*)(ws + 48 * MB);           // 16 MB fp32

  cvt_bf16<<<(4096 * 1024) / 256, 256, 0, stream>>>(x, xb, 4096 * 1024);
  cvt_w_t<<<(1024 * 1024) / 256, 256, 0, stream>>>(Wq, WqT);
  cvt_w_t<<<(1024 * 1024) / 256, 256, 0, stream>>>(Wk, WkT);
  cvt_w_t<<<(1024 * 1024) / 256, 256, 0, stream>>>(Wv, WvT);
  cvt_w_t<<<(1024 * 1024) / 256, 256, 0, stream>>>(Wp, WpT);

  dim3 gg(8, 32);  // N/128 x M/128
  gemm_bf16<<<gg, 256, 0, stream>>>(xb, WqT, bq, nullptr, Qh, nullptr, 0);
  gemm_bf16<<<gg, 256, 0, stream>>>(xb, WkT, bk, nullptr, Kh, nullptr, 0);
  gemm_bf16<<<gg, 256, 0, stream>>>(xb, WvT, bv, nullptr, Vt, nullptr, 1);

  attention_fa<<<dim3(16, 32), 256, 0, stream>>>(Qh, Kh, Vt, AO);

  gemm_bf16<<<gg, 256, 0, stream>>>(AO, WpT, bp, x, nullptr, Yt, 2);

  layernorm_row<<<4096, 256, 0, stream>>>(Yt, gamma, beta, out);
}
